// GCN_Protein_53807350284438
// MI455X (gfx1250) — matrix operand order repaired and checked by reading, not compiled, not run on hardware
//
#include <hip/hip_runtime.h>
#include <hip/hip_bf16.h>
#include <math.h>

// ---------------------------------------------------------------------------
// GCN (3x GCNConv + mean-pool + MLP head) for MI455X / gfx1250.
// Dense GEMMs use v_wmma_f32_16x16x32_bf16 (f32 accumulate); graph
// aggregation uses native f32 global atomics (L2-resident: 51.2MB << 192MB).
// ---------------------------------------------------------------------------

#define N_NODES  100000
#define N_EDGES  1600000
#define N_GRAPHS 512
#define M_TILES  (N_NODES / 16)   // 6250, exact

typedef __attribute__((ext_vector_type(16))) __bf16 v16bf;
typedef __attribute__((ext_vector_type(8)))  float  v8f;

union BFrag { v16bf v; unsigned short u[16]; };

__device__ __forceinline__ unsigned short f2bf(float f) {
    // round-to-nearest-even f32 -> bf16
    unsigned int x = __float_as_uint(f);
    unsigned int r = x + 0x7FFFu + ((x >> 16) & 1u);
    return (unsigned short)(r >> 16);
}

__device__ __forceinline__ void atomAddF(float* p, float v) {
    __hip_atomic_fetch_add(p, v, __ATOMIC_RELAXED, __HIP_MEMORY_SCOPE_AGENT);
}

// ------------------------------ degree / norm ------------------------------

__global__ void init_deg_kernel(float* deg, int n) {
    int i = blockIdx.x * blockDim.x + threadIdx.x;
    if (i < n) deg[i] = 1.0f;                       // self-loop
}

__global__ void deg_accum_kernel(const int* __restrict__ dst, float* deg, int e) {
    int i = blockIdx.x * blockDim.x + threadIdx.x;
    if (i < e) atomAddF(&deg[dst[i]], 1.0f);
}

__global__ void dinv_kernel(const float* __restrict__ deg, float* dinv, int n) {
    int i = blockIdx.x * blockDim.x + threadIdx.x;
    if (i < n) dinv[i] = rsqrtf(deg[i]);            // deg >= 1 always
}

// --------------------------- weight packing (B) ----------------------------
// B-fragment layout (bf16 16x16x32): lane L covers column n = nt*16 + (L&15);
// its 16 elements cover k = kt*32 + (j>>3)*16 + (L>>4)*8 + (j&7).  Packed so each lane's
// fragment is 32 contiguous bytes -> two b128 loads in the GEMM.

__global__ void pack_w_kernel(const float* __restrict__ W, unsigned short* __restrict__ Wp,
                              int K, int F) {
    int tid = blockIdx.x * blockDim.x + threadIdx.x;
    if (tid >= K * F) return;
    int j    = tid & 15;
    int lane = (tid >> 4) & 31;
    int rest = tid >> 9;
    int NT   = F >> 4;
    int nt   = rest % NT;
    int kt   = rest / NT;
    int k    = kt * 32 + (j >> 3) * 16 + (lane >> 4) * 8 + (j & 7);
    int n    = nt * 16 + (lane & 15);
    Wp[tid]  = f2bf(W[k * F + n]);
}

// ------------------------------ WMMA GEMM ----------------------------------
// h[N,F] = act[N,K] @ W[K,F].  One 16-row M-tile per wave; A fragments built
// inline (f32 -> bf16), all K/32 fragments held in registers; loop N-tiles.

template <int K, int F>
__global__ __launch_bounds__(256)
void gemm_bf16_kernel(const float* __restrict__ act,
                      const unsigned short* __restrict__ Wp,
                      float* __restrict__ out, int mtiles) {
    constexpr int KT = K / 32;
    constexpr int NT = F / 16;
    const int lane = threadIdx.x & 31;
    const int wave = threadIdx.x >> 5;
    const int mt   = blockIdx.x * 8 + wave;
    if (mt >= mtiles) return;                       // wave-uniform: EXEC stays full

    const int g = lane >> 4;
    const int m = mt * 16 + (lane & 15);
    const float* rowp = act + (size_t)m * K;

    BFrag a[KT];
#pragma unroll
    for (int kt = 0; kt < KT; ++kt) {
        // j=0..7  -> k = kt*32 + g*8 + j        (contiguous 8 floats)
        // j=8..15 -> k = kt*32 + 16 + g*8 + j-8 (contiguous 8 floats)
        const float* p0 = rowp + kt * 32 + g * 8;
        float4 f0 = *(const float4*)(p0);
        float4 f1 = *(const float4*)(p0 + 4);
        float4 f2 = *(const float4*)(p0 + 16);
        float4 f3 = *(const float4*)(p0 + 20);
        a[kt].u[0]  = f2bf(f0.x); a[kt].u[1]  = f2bf(f0.y);
        a[kt].u[2]  = f2bf(f0.z); a[kt].u[3]  = f2bf(f0.w);
        a[kt].u[4]  = f2bf(f1.x); a[kt].u[5]  = f2bf(f1.y);
        a[kt].u[6]  = f2bf(f1.z); a[kt].u[7]  = f2bf(f1.w);
        a[kt].u[8]  = f2bf(f2.x); a[kt].u[9]  = f2bf(f2.y);
        a[kt].u[10] = f2bf(f2.z); a[kt].u[11] = f2bf(f2.w);
        a[kt].u[12] = f2bf(f3.x); a[kt].u[13] = f2bf(f3.y);
        a[kt].u[14] = f2bf(f3.z); a[kt].u[15] = f2bf(f3.w);
    }

#pragma unroll
    for (int nt = 0; nt < NT; ++nt) {
        v8f c = {0.f, 0.f, 0.f, 0.f, 0.f, 0.f, 0.f, 0.f};
#pragma unroll
        for (int kt = 0; kt < KT; ++kt) {
            BFrag b;
            b.v = *(const v16bf*)(Wp + ((size_t)(kt * NT + nt) * 32 + lane) * 16);
            c = __builtin_amdgcn_wmma_f32_16x16x32_bf16(
                    /*neg_a=*/false, a[kt].v, /*neg_b=*/false, b.v,
                    /*c_mod=*/(short)0, c, /*reuse_a=*/false, /*reuse_b=*/false);
        }
        // C/D layout: VGPR r -> row = mt*16 + g*8 + r, col = nt*16 + (lane&15)
        float* op = out + (size_t)(mt * 16 + g * 8) * F + nt * 16 + (lane & 15);
#pragma unroll
        for (int r = 0; r < 8; ++r) op[(size_t)r * F] = c[r];
    }
}

// ----------------------------- aggregation ---------------------------------

// agg[i,f] = dinv[i]^2 * h[i,f]   (self-loop term; also zero-initializes agg)
__global__ void self_init_kernel(const float* __restrict__ h, const float* __restrict__ dinv,
                                 float* __restrict__ agg, int total, int shift) {
    int tid = blockIdx.x * blockDim.x + threadIdx.x;
    if (tid >= total) return;
    int i = tid >> shift;
    float di = dinv[i];
    agg[tid] = di * di * h[tid];
}

// agg[dst] += dinv[src]*dinv[dst] * h[src]   (float4 per thread)
template <int F>
__global__ void edge_scatter_kernel(const int* __restrict__ src, const int* __restrict__ dst,
                                    const float* __restrict__ dinv, const float* __restrict__ h,
                                    float* __restrict__ agg, int e) {
    constexpr int C = F / 4;
    int tid = blockIdx.x * blockDim.x + threadIdx.x;
    if (tid >= e * C) return;
    int ei = tid / C;
    int c  = tid - ei * C;
    int s  = src[ei];
    int d  = dst[ei];
    float w = dinv[s] * dinv[d];
    float4 v = *(const float4*)(h + (size_t)s * F + c * 4);
    float* o = agg + (size_t)d * F + c * 4;
    atomAddF(o + 0, w * v.x);
    atomAddF(o + 1, w * v.y);
    atomAddF(o + 2, w * v.z);
    atomAddF(o + 3, w * v.w);
}

__global__ void bias_relu_kernel(float* __restrict__ agg, const float* __restrict__ bias,
                                 int total, int mask) {
    int tid = blockIdx.x * blockDim.x + threadIdx.x;
    if (tid >= total) return;
    agg[tid] = fmaxf(agg[tid] + bias[tid & mask], 0.0f);
}

// -------------------------------- pooling ----------------------------------

__global__ void zero_kernel(float* p, int n) {
    int i = blockIdx.x * blockDim.x + threadIdx.x;
    if (i < n) p[i] = 0.0f;
}

__global__ void pool_accum_kernel(const float* __restrict__ act, const int* __restrict__ batch,
                                  float* __restrict__ pool, float* __restrict__ cnt, int n) {
    int tid = blockIdx.x * blockDim.x + threadIdx.x;
    if (tid >= n * 128) return;
    int i = tid >> 7;
    int f = tid & 127;
    int b = batch[i];
    atomAddF(&pool[b * 128 + f], act[tid]);
    if (f == 0) atomAddF(&cnt[b], 1.0f);
}

// ------------------------------ classifier ---------------------------------

__global__ void classifier_kernel(const float* __restrict__ pool, const float* __restrict__ cnt,
                                  const float* __restrict__ Wc1, const float* __restrict__ bc1,
                                  const float* __restrict__ Wc2, const float* __restrict__ bc2,
                                  float* __restrict__ out) {
    __shared__ float p[128];
    __shared__ float z[64];
    __shared__ float lg[10];
    __shared__ float red[2];
    int g = blockIdx.x, t = threadIdx.x;
    float ic = 1.0f / fmaxf(cnt[g], 1.0f);
    if (t < 128) p[t] = pool[g * 128 + t] * ic;
    __syncthreads();
    if (t < 64) {
        float acc = bc1[t];
        for (int k = 0; k < 128; ++k) acc += p[k] * Wc1[k * 64 + t];
        z[t] = fmaxf(acc, 0.0f);
    }
    __syncthreads();
    if (t < 10) {
        float acc = bc2[t];
        for (int k = 0; k < 64; ++k) acc += z[k] * Wc2[k * 10 + t];
        lg[t] = acc;
    }
    __syncthreads();
    if (t == 0) {
        float m = lg[0];
        for (int o = 1; o < 10; ++o) m = fmaxf(m, lg[o]);
        float s = 0.0f;
        for (int o = 0; o < 10; ++o) s += expf(lg[o] - m);
        red[0] = m;
        red[1] = logf(s);
    }
    __syncthreads();
    if (t < 10) out[g * 10 + t] = lg[t] - red[0] - red[1];
}

// ------------------------------- launcher ----------------------------------

extern "C" void kernel_launch(void* const* d_in, const int* in_sizes, int n_in,
                              void* d_out, int out_size, void* d_ws, size_t ws_size,
                              hipStream_t stream) {
    const int N = N_NODES, E = N_EDGES;

    const float* x   = (const float*)d_in[0];
    const int*   ei  = (const int*)d_in[1];       // [2,E]: row0=src, row1=dst
    const int*   bat = (const int*)d_in[2];
    const float* W1  = (const float*)d_in[3];
    const float* b1  = (const float*)d_in[4];
    const float* W2  = (const float*)d_in[5];
    const float* b2  = (const float*)d_in[6];
    const float* W3  = (const float*)d_in[7];
    const float* b3  = (const float*)d_in[8];
    const float* Wc1 = (const float*)d_in[9];
    const float* bc1 = (const float*)d_in[10];
    const float* Wc2 = (const float*)d_in[11];
    const float* bc2 = (const float*)d_in[12];

    const int* src = ei;
    const int* dst = ei + E;

    // workspace layout (~155 MB)
    char* w = (char*)d_ws;
    float* deg  = (float*)w; w += (size_t)N * 4;
    float* dinv = (float*)w; w += (size_t)N * 4;
    float* bufA = (float*)w; w += (size_t)N * 128 * 4;
    float* bufB = (float*)w; w += (size_t)N * 128 * 4;
    float* bufH = (float*)w; w += (size_t)N * 128 * 4;
    unsigned short* Wp1 = (unsigned short*)w; w += 128 * 64 * 2;
    unsigned short* Wp2 = (unsigned short*)w; w += 64 * 128 * 2;
    unsigned short* Wp3 = (unsigned short*)w; w += 128 * 128 * 2;
    float* pool = (float*)w; w += (size_t)N_GRAPHS * 128 * 4;
    float* cnt  = (float*)w; w += (size_t)N_GRAPHS * 4;

    const int TB = 256;
    const int gemm_grid = (M_TILES + 7) / 8;   // 8 waves (16-row tiles) per block

    // ---- normalization ----
    init_deg_kernel<<<(N + TB - 1) / TB, TB, 0, stream>>>(deg, N);
    deg_accum_kernel<<<(E + TB - 1) / TB, TB, 0, stream>>>(dst, deg, E);
    dinv_kernel<<<(N + TB - 1) / TB, TB, 0, stream>>>(deg, dinv, N);

    // ---- weight packing ----
    pack_w_kernel<<<(128 * 64 + TB - 1) / TB, TB, 0, stream>>>(W1, Wp1, 128, 64);
    pack_w_kernel<<<(64 * 128 + TB - 1) / TB, TB, 0, stream>>>(W2, Wp2, 64, 128);
    pack_w_kernel<<<(128 * 128 + TB - 1) / TB, TB, 0, stream>>>(W3, Wp3, 128, 128);

    // ---- layer 1: 128 -> 64 ----
    gemm_bf16_kernel<128, 64><<<gemm_grid, TB, 0, stream>>>(x, Wp1, bufH, M_TILES);
    self_init_kernel<<<(N * 64 + TB - 1) / TB, TB, 0, stream>>>(bufH, dinv, bufA, N * 64, 6);
    edge_scatter_kernel<64><<<(E * 16 + TB - 1) / TB, TB, 0, stream>>>(src, dst, dinv, bufH, bufA, E);
    bias_relu_kernel<<<(N * 64 + TB - 1) / TB, TB, 0, stream>>>(bufA, b1, N * 64, 63);

    // ---- layer 2: 64 -> 128 ----
    gemm_bf16_kernel<64, 128><<<gemm_grid, TB, 0, stream>>>(bufA, Wp2, bufH, M_TILES);
    self_init_kernel<<<(N * 128 + TB - 1) / TB, TB, 0, stream>>>(bufH, dinv, bufB, N * 128, 7);
    edge_scatter_kernel<128><<<(E * 32 + TB - 1) / TB, TB, 0, stream>>>(src, dst, dinv, bufH, bufB, E);
    bias_relu_kernel<<<(N * 128 + TB - 1) / TB, TB, 0, stream>>>(bufB, b2, N * 128, 127);

    // ---- layer 3: 128 -> 128 ----
    gemm_bf16_kernel<128, 128><<<gemm_grid, TB, 0, stream>>>(bufB, Wp3, bufH, M_TILES);
    self_init_kernel<<<(N * 128 + TB - 1) / TB, TB, 0, stream>>>(bufH, dinv, bufA, N * 128, 7);
    edge_scatter_kernel<128><<<(E * 32 + TB - 1) / TB, TB, 0, stream>>>(src, dst, dinv, bufH, bufA, E);
    bias_relu_kernel<<<(N * 128 + TB - 1) / TB, TB, 0, stream>>>(bufA, b3, N * 128, 127);

    // ---- global mean pool ----
    zero_kernel<<<(N_GRAPHS * 128 + N_GRAPHS + TB - 1) / TB, TB, 0, stream>>>(pool, N_GRAPHS * 128 + N_GRAPHS);
    pool_accum_kernel<<<(N * 128 + TB - 1) / TB, TB, 0, stream>>>(bufA, bat, pool, cnt, N);

    // ---- classifier + log_softmax ----
    classifier_kernel<<<N_GRAPHS, 128, 0, stream>>>(pool, cnt, Wc1, bc1, Wc2, bc2, (float*)d_out);
}
